// Block_69681549410915
// MI455X (gfx1250) — matrix operand order repaired and checked by reading, not compiled, not run on hardware
//
#include <hip/hip_runtime.h>

// ---------------------------------------------------------------- constants
#define BB      2
#define SS      1024
#define HHD     2048
#define NHEADS  16
#define QCC     1536
#define KVCC    512
#define ROPED   64
#define NOPED   128
#define VHD     128
#define INTERD  1408
#define NSHE    2
#define NRTE    8
#define TT      (BB*SS)
#define QDIM    (NOPED+ROPED)   // 192
#define QPROJ   (NHEADS*QDIM)   // 3072
#define KVD     (NOPED+VHD)     // 256
#define KVPROJ  (NHEADS*KVD)    // 4096
#define CKVD    (KVCC+ROPED)    // 576
#define EPSF    1e-6f

#define USE_ASYNC 1
#define USE_TDM   1

typedef __attribute__((ext_vector_type(16))) __bf16       v16bf;
typedef __attribute__((ext_vector_type(8)))  float        v8f;
typedef __attribute__((ext_vector_type(8)))  unsigned int v8u;
typedef __attribute__((ext_vector_type(4)))  int          v4i;
typedef __attribute__((ext_vector_type(8)))  int          v8i;

static __device__ __forceinline__ unsigned short f2bf(float f) {
  unsigned u = __builtin_bit_cast(unsigned, f);
  u += 0x7FFFu + ((u >> 16) & 1u);                 // RNE
  return (unsigned short)(u >> 16);
}
static __device__ __forceinline__ float bf2f(unsigned short h) {
  return __builtin_bit_cast(float, (unsigned)h << 16);
}

#if USE_ASYNC
static __device__ __forceinline__ void async16(unsigned lds_addr, const unsigned short* g) {
  asm volatile("global_load_async_to_lds_b128 %0, %1, off" :: "v"(lds_addr), "v"(g) : "memory");
}
#if __has_builtin(__builtin_amdgcn_s_wait_asynccnt)
#define WAIT_ASYNC() __builtin_amdgcn_s_wait_asynccnt(0)
#else
#define WAIT_ASYNC() asm volatile("s_wait_asynccnt 0x0" ::: "memory")
#endif
#else
#define WAIT_ASYNC()
#endif

#if USE_TDM
static __device__ __forceinline__ void tdm_load(v4i g0, v8i g1) {
  // TENSOR_LOAD_TO_LDS: D# group0 (4 SGPRs), group1 (8 SGPRs); 2D tensor -> groups 2/3 NULL
  asm volatile("tensor_load_to_lds %0, %1" :: "s"(g0), "s"(g1) : "memory");
}
#if __has_builtin(__builtin_amdgcn_s_wait_tensorcnt)
#define WAIT_TENSOR() __builtin_amdgcn_s_wait_tensorcnt(0)
#else
#define WAIT_TENSOR() asm volatile("s_wait_tensorcnt 0x0" ::: "memory")
#endif
#endif

// ------------------------------------------------------------ WMMA GEMM (pure bf16)
// C[M,N] (+)= A[M,K] * B^T where A: bf16 [M][K] (lda), B: bf16 [N][K] (ldb).
// 256 thr = 8 waves (4 M x 2 N); block tile 128x64; each wave owns 32x32 C (4 WMMA accs).
// K-stage 64, double-buffered LDS. A staged by per-lane async copies (all waves);
// B staged by TDM 2D descriptor (wave 0), LDS row pad expressed via D#.pad_*.
#define ROWB   144                  // LDS row pitch bytes: 128B data + 16B pad
#define STAGEA (128*ROWB)           // A stage: 18432 B
#define STAGEB (64*ROWB)            // B stage:  9216 B

template<bool OUTBF, bool ACCUM>
__global__ __launch_bounds__(256)
void gemm_bb(const unsigned short* __restrict__ A, int lda,
             const unsigned short* __restrict__ B, int ldb,
             void* __restrict__ Cp, int ldc,
             int M, int N, int K)
{
  __shared__ alignas(16) unsigned short As[2][128][72];
  __shared__ alignas(16) unsigned short Bs[2][64][72];

  const int tid  = threadIdx.x;
  const int lane = tid & 31;
  const int wave = tid >> 5;
  const int wm   = wave >> 1;         // 0..3 -> 32 rows each
  const int wn   = wave & 1;          // 0..1 -> 32 cols each
  const int half = lane >> 4;
  const int r    = lane & 15;

  const int tileM = blockIdx.y * 128;
  const int tileN = blockIdx.x * 64;

  const unsigned ldsA = (unsigned)(size_t)(&As[0][0][0]);
  const unsigned ldsB = (unsigned)(size_t)(&Bs[0][0][0]);

  v8f acc00 = {}, acc01 = {}, acc10 = {}, acc11 = {};

  // A staging: 128 rows x 128B; thread -> 64B (4 x b128)
  const int arows = tid >> 1;           // 0..127
  const int acol  = (tid & 1) * 32;     // element offset (32 bf16 = 64B)
  // B staging fallback: 64 rows x 128B; thread -> 32B (2 x b128)
  const int brows = tid >> 2;           // 0..63
  const int bcol  = (tid & 3) * 16;

  auto stage = [&](int buf, int k0) {
#if USE_ASYNC
    const unsigned short* ga = A + (size_t)(tileM + arows) * lda + k0 + acol;
    unsigned la = ldsA + (unsigned)buf * STAGEA + arows * ROWB + acol * 2;
    async16(la,      ga);
    async16(la + 16, ga + 8);
    async16(la + 32, ga + 16);
    async16(la + 48, ga + 24);
#if USE_TDM
    if (wave == 0) {
      unsigned long long gaddr =
          (unsigned long long)(size_t)(B + (size_t)tileN * ldb + k0);
      v4i g0;
      g0[0] = 1;                                            // count=1 (valid D#)
      g0[1] = (int)(ldsB + (unsigned)buf * STAGEB);         // lds_addr
      g0[2] = (int)(unsigned)gaddr;                         // global_addr[31:0]
      g0[3] = (int)(((gaddr >> 32) & 0x01FFFFFFu) | (2u << 30)); // addr[56:32] | type=2
      v8i g1;
      g1[0] = (1 << 16) | (1 << 20) | (4 << 22) | (3 << 25);
      //        data=2B   pad_en      every 32DW   pad 4DW (-> 144B row pitch)
      g1[1] = 0;                                            // tensor_dim0 low16 << 16 (2^30)
      g1[2] = 0x4000;                                       // dim0 hi16 | dim1 low16<<16
      g1[3] = 0x4000 | (64 << 16);                          // dim1 hi16 | tile_dim0=64
      g1[4] = 64;                                           // tile_dim1=64 | tile_dim2=0
      g1[5] = ldb;                                          // tensor_dim0_stride[31:0]
      g1[6] = 0;                                            // stride hi16 | dim1_stride low
      g1[7] = 0;
      tdm_load(g0, g1);
    }
#else
    const unsigned short* gb = B + (size_t)(tileN + brows) * ldb + k0 + bcol;
    unsigned lb = ldsB + (unsigned)buf * STAGEB + brows * ROWB + bcol * 2;
    async16(lb,      gb);
    async16(lb + 16, gb + 8);
#endif
#else
    const unsigned short* ga = A + (size_t)(tileM + arows) * lda + k0 + acol;
    *(uint4*)(&As[buf][arows][acol])      = *(const uint4*)ga;
    *(uint4*)(&As[buf][arows][acol + 8])  = *(const uint4*)(ga + 8);
    *(uint4*)(&As[buf][arows][acol + 16]) = *(const uint4*)(ga + 16);
    *(uint4*)(&As[buf][arows][acol + 24]) = *(const uint4*)(ga + 24);
    const unsigned short* gb = B + (size_t)(tileN + brows) * ldb + k0 + bcol;
    *(uint4*)(&Bs[buf][brows][bcol])     = *(const uint4*)gb;
    *(uint4*)(&Bs[buf][brows][bcol + 8]) = *(const uint4*)(gb + 8);
#endif
  };

  auto sync_stage = [&]() {
    WAIT_ASYNC();
#if USE_ASYNC && USE_TDM
    if (wave == 0) { WAIT_TENSOR(); }
#endif
    __syncthreads();
  };

  stage(0, 0);
  sync_stage();

  int buf = 0;
  for (int k0 = 0; k0 < K; k0 += 64) {
    if (k0 + 64 < K) stage(buf ^ 1, k0 + 64);

    const unsigned* rowA0 = (const unsigned*)(&As[buf][wm * 32 + r][0]);
    const unsigned* rowA1 = (const unsigned*)(&As[buf][wm * 32 + 16 + r][0]);
    const unsigned* rowB0 = (const unsigned*)(&Bs[buf][wn * 32 + r][0]);
    const unsigned* rowB1 = (const unsigned*)(&Bs[buf][wn * 32 + 16 + r][0]);

#pragma unroll
    for (int kk = 0; kk < 2; ++kk) {
      v8u au0, au1, bu0, bu1;
#pragma unroll
      for (int v = 0; v < 8; ++v) {
        int kpA = kk * 16 + ((v < 4 ? v : 8 + (v - 4)) + 4 * half);
        au0[v] = rowA0[kpA];
        au1[v] = rowA1[kpA];
        int kpB = kk * 16 + ((v < 4 ? v : 8 + (v - 4)) + 4 * half);
        bu0[v] = rowB0[kpB];
        bu1[v] = rowB1[kpB];
      }
      v16bf a0 = __builtin_bit_cast(v16bf, au0);
      v16bf a1 = __builtin_bit_cast(v16bf, au1);
      v16bf b0 = __builtin_bit_cast(v16bf, bu0);
      v16bf b1 = __builtin_bit_cast(v16bf, bu1);
      acc00 = __builtin_amdgcn_wmma_f32_16x16x32_bf16(false, a0, false, b0, (short)0, acc00, false, false);
      acc01 = __builtin_amdgcn_wmma_f32_16x16x32_bf16(false, a0, false, b1, (short)0, acc01, false, false);
      acc10 = __builtin_amdgcn_wmma_f32_16x16x32_bf16(false, a1, false, b0, (short)0, acc10, false, false);
      acc11 = __builtin_amdgcn_wmma_f32_16x16x32_bf16(false, a1, false, b1, (short)0, acc11, false, false);
    }
    sync_stage();
    buf ^= 1;
  }

  float*          Cf = (float*)Cp;
  unsigned short* Cb = (unsigned short*)Cp;
#pragma unroll
  for (int v = 0; v < 8; ++v) {
    int  row0 = tileM + wm * 32 + v + 8 * half;
    int  row1 = row0 + 16;
    long b0 = (long)row0 * ldc + tileN + wn * 32 + r;
    long b1 = (long)row1 * ldc + tileN + wn * 32 + r;
    if (OUTBF) {
      Cb[b0] = f2bf(acc00[v]); Cb[b0 + 16] = f2bf(acc01[v]);
      Cb[b1] = f2bf(acc10[v]); Cb[b1 + 16] = f2bf(acc11[v]);
    } else if (ACCUM) {
      Cf[b0] += acc00[v]; Cf[b0 + 16] += acc01[v];
      Cf[b1] += acc10[v]; Cf[b1 + 16] += acc11[v];
    } else {
      Cf[b0] = acc00[v]; Cf[b0 + 16] = acc01[v];
      Cf[b1] = acc10[v]; Cf[b1 + 16] = acc11[v];
    }
  }
}

// ------------------------------------------------------------ weight transpose+convert
// in: fp32 [K][N] (+z*K*N)  ->  out: bf16 [N][K] (+z*K*N)
__global__ __launch_bounds__(256)
void transpose_w_kernel(const float* __restrict__ in, unsigned short* __restrict__ out,
                        int K, int N)
{
  __shared__ unsigned short t[32][33];
  long zo = (long)blockIdx.z * K * N;
  const float* ip = in + zo;
  unsigned short* op = out + zo;
  int kb = blockIdx.y * 32, nb = blockIdx.x * 32;
  int tx = threadIdx.x & 31, ty = threadIdx.x >> 5;     // 32 x 8
#pragma unroll
  for (int i = 0; i < 32; i += 8)
    t[ty + i][tx] = f2bf(ip[(long)(kb + ty + i) * N + nb + tx]);
  __syncthreads();
#pragma unroll
  for (int i = 0; i < 32; i += 8)
    op[(long)(nb + ty + i) * K + kb + tx] = t[tx][ty + i];
}

// v^T builder: kvflat_bf [b,t, h*256+128+d] -> vT [ (b*16+h), d, t ]
__global__ __launch_bounds__(256)
void vtranspose_kernel(const unsigned short* __restrict__ kv, unsigned short* __restrict__ vt)
{
  __shared__ unsigned short t[32][33];
  int z = blockIdx.z, b = z >> 4, h = z & 15;
  const unsigned short* ip = kv + (size_t)b * SS * KVPROJ + h * KVD + NOPED;
  unsigned short* op = vt + (size_t)z * VHD * SS;
  int tb = blockIdx.y * 32, db = blockIdx.x * 32;
  int tx = threadIdx.x & 31, ty = threadIdx.x >> 5;
#pragma unroll
  for (int i = 0; i < 32; i += 8)
    t[ty + i][tx] = ip[(size_t)(tb + ty + i) * KVPROJ + db + tx];
  __syncthreads();
#pragma unroll
  for (int i = 0; i < 32; i += 8)
    op[(size_t)(db + ty + i) * SS + tb + tx] = t[tx][ty + i];
}

// ------------------------------------------------------------ RMSNorm (fp32 in -> bf16 out)
__global__ __launch_bounds__(256)
void rmsnorm_kernel(const float* __restrict__ in, int istride,
                    const float* __restrict__ w,
                    unsigned short* __restrict__ out, int ostride, int W)
{
  __shared__ float red[256];
  const long row = blockIdx.x;
  const float* p = in + row * (long)istride;
  unsigned short* q = out + row * (long)ostride;
  float ss = 0.f;
  for (int j = threadIdx.x; j < W; j += 256) { float v = p[j]; ss += v * v; }
  red[threadIdx.x] = ss; __syncthreads();
  for (int s = 128; s > 0; s >>= 1) {
    if (threadIdx.x < s) red[threadIdx.x] += red[threadIdx.x + s];
    __syncthreads();
  }
  const float scale = rsqrtf(red[0] / (float)W + EPSF);
  for (int j = threadIdx.x; j < W; j += 256) q[j] = f2bf(p[j] * scale * w[j]);
}

// ------------------------------------------------------------ RoPE
__global__ void rope_q_kernel(unsigned short* __restrict__ q, const float* __restrict__ freq)
{
  int idx = blockIdx.x * blockDim.x + threadIdx.x;   // TT*16*32 exactly
  int i = idx & 31, h = (idx >> 5) & 15, t = idx >> 9;
  int s = t % SS;
  float f = freq[s * 32 + i], c = cosf(f), sn = sinf(f);
  unsigned short* p = q + (size_t)t * QPROJ + h * QDIM + NOPED + 2 * i;
  float x1 = bf2f(p[0]), x2 = bf2f(p[1]);
  p[0] = f2bf(x1 * c - x2 * sn);
  p[1] = f2bf(x1 * sn + x2 * c);
}

__global__ void rope_k_kernel(const float* __restrict__ ckv, unsigned short* __restrict__ kr,
                              const float* __restrict__ freq)
{
  int idx = blockIdx.x * blockDim.x + threadIdx.x;   // TT*32 exactly
  int i = idx & 31, t = idx >> 5;
  int s = t % SS;
  float f = freq[s * 32 + i], c = cosf(f), sn = sinf(f);
  const float* sp = ckv + (size_t)t * CKVD + KVCC + 2 * i;
  unsigned short* dp = kr + (size_t)t * ROPED + 2 * i;
  float x1 = sp[0], x2 = sp[1];
  dp[0] = f2bf(x1 * c - x2 * sn);
  dp[1] = f2bf(x1 * sn + x2 * c);
}

// ------------------------------------------------------------ softmax rows (fp32 -> bf16)
__global__ __launch_bounds__(256)
void softmax_kernel(float* __restrict__ sc, const float* __restrict__ mask,
                    unsigned short* __restrict__ ob, int W, float scale)
{
  __shared__ float red[256];
  const long row = blockIdx.x;
  const int  s   = (int)(row % SS);
  float* p = sc + row * (long)W;
  unsigned short* q = ob + row * (long)W;
  const float* m = mask + (long)s * W;

  float mx = -3.0e38f;
  for (int j = threadIdx.x; j < W; j += 256) mx = fmaxf(mx, p[j] * scale + m[j]);
  red[threadIdx.x] = mx; __syncthreads();
  for (int st = 128; st > 0; st >>= 1) {
    if (threadIdx.x < st) red[threadIdx.x] = fmaxf(red[threadIdx.x], red[threadIdx.x + st]);
    __syncthreads();
  }
  mx = red[0]; __syncthreads();

  float sum = 0.f;
  for (int j = threadIdx.x; j < W; j += 256) {
    float v = expf(p[j] * scale + m[j] - mx);
    p[j] = v; sum += v;
  }
  red[threadIdx.x] = sum; __syncthreads();
  for (int st = 128; st > 0; st >>= 1) {
    if (threadIdx.x < st) red[threadIdx.x] += red[threadIdx.x + st];
    __syncthreads();
  }
  const float inv = 1.f / red[0];
  for (int j = threadIdx.x; j < W; j += 256) q[j] = f2bf(p[j] * inv);
}

// ------------------------------------------------------------ elementwise
__global__ void add_kernel(const float* __restrict__ a, const float* __restrict__ b,
                           float* __restrict__ o, long n)
{
  long i = (long)blockIdx.x * blockDim.x + threadIdx.x;
  if (i < n) o[i] = a[i] + b[i];
}

__global__ void silu_mul_kernel(const float* __restrict__ g, const float* __restrict__ u,
                                unsigned short* __restrict__ o, long n)
{
  long i = (long)blockIdx.x * blockDim.x + threadIdx.x;
  if (i < n) { float x = g[i]; o[i] = f2bf((x / (1.f + expf(-x))) * u[i]); }
}

__global__ void axpy_rows_kernel(float* __restrict__ acc, const float* __restrict__ t,
                                 const float* __restrict__ w, int e, long n, int width)
{
  long i = (long)blockIdx.x * blockDim.x + threadIdx.x;
  if (i < n) acc[i] += w[(i / width) * NRTE + e] * t[i];
}

// ------------------------------------------------------------ router (softmax + top-2)
__global__ __launch_bounds__(256)
void router_kernel(const unsigned short* __restrict__ t, const float* __restrict__ Wr,
                   float* __restrict__ wout)
{
  __shared__ float sacc[256][NRTE];
  const int tok = blockIdx.x;
  float acc[NRTE];
#pragma unroll
  for (int e = 0; e < NRTE; ++e) acc[e] = 0.f;
  for (int h = threadIdx.x; h < HHD; h += 256) {
    float tv = bf2f(t[(size_t)tok * HHD + h]);
#pragma unroll
    for (int e = 0; e < NRTE; ++e) acc[e] += tv * Wr[h * NRTE + e];
  }
#pragma unroll
  for (int e = 0; e < NRTE; ++e) sacc[threadIdx.x][e] = acc[e];
  __syncthreads();
  for (int s = 128; s > 0; s >>= 1) {
    if (threadIdx.x < s)
      for (int e = 0; e < NRTE; ++e) sacc[threadIdx.x][e] += sacc[threadIdx.x + s][e];
    __syncthreads();
  }
  if (threadIdx.x == 0) {
    float p[NRTE]; float mx = -3.0e38f;
    for (int e = 0; e < NRTE; ++e) mx = fmaxf(mx, sacc[0][e]);
    float sum = 0.f;
    for (int e = 0; e < NRTE; ++e) { p[e] = expf(sacc[0][e] - mx); sum += p[e]; }
    for (int e = 0; e < NRTE; ++e) p[e] /= sum;
    int i0 = 0;
    for (int e = 1; e < NRTE; ++e) if (p[e] > p[i0]) i0 = e;
    int i1 = -1;
    for (int e = 0; e < NRTE; ++e) if (e != i0 && (i1 < 0 || p[e] > p[i1])) i1 = e;
    for (int e = 0; e < NRTE; ++e)
      wout[(size_t)tok * NRTE + e] = (e == i0) ? p[i0] : (e == i1) ? p[i1] : 0.f;
  }
}

// ------------------------------------------------------------ host orchestration
static inline long cdiv(long a, long b) { return (a + b - 1) / b; }

extern "C" void kernel_launch(void* const* d_in, const int* in_sizes, int n_in,
                              void* d_out, int out_size, void* d_ws, size_t ws_size,
                              hipStream_t stream)
{
  (void)in_sizes; (void)n_in; (void)out_size; (void)ws_size;

  const float* x         = (const float*)d_in[0];
  const float* freq      = (const float*)d_in[1];
  const float* mask      = (const float*)d_in[2];
  const float* w_attn    = (const float*)d_in[3];
  const float* w_ffn     = (const float*)d_in[4];
  const float* W_dq      = (const float*)d_in[5];
  const float* q_c_norm  = (const float*)d_in[6];
  const float* W_uq      = (const float*)d_in[7];
  const float* W_dkv     = (const float*)d_in[8];
  const float* kv_c_norm = (const float*)d_in[9];
  const float* W_ukv     = (const float*)d_in[10];
  const float* W_o       = (const float*)d_in[11];
  const float* W_r       = (const float*)d_in[12];
  const float* Ws_gate   = (const float*)d_in[13];
  const float* Ws_up     = (const float*)d_in[14];
  const float* Ws_down   = (const float*)d_in[15];
  const float* We_gate   = (const float*)d_in[16];
  const float* We_up     = (const float*)d_in[17];
  const float* We_down   = (const float*)d_in[18];
  float* out = (float*)d_out;

  // ---- workspace carve-up
  char* wsb = (char*)d_ws;
  size_t off = 0;
  auto allocf = [&](size_t n) { float* p = (float*)(wsb + off); off += n * 4; return p; };
  auto allocb = [&](size_t n) { unsigned short* p = (unsigned short*)(wsb + off);
                                off += n * 2; off = (off + 15) & ~(size_t)15; return p; };
  // bf16 transposed weights [N][K]
  unsigned short* WdqT  = allocb((size_t)HHD * QCC);
  unsigned short* WuqT  = allocb((size_t)QCC * QPROJ);
  unsigned short* WdkvT = allocb((size_t)HHD * CKVD);
  unsigned short* WukvT = allocb((size_t)KVCC * KVPROJ);
  unsigned short* WoT   = allocb((size_t)HHD * HHD);
  unsigned short* WsgT  = allocb((size_t)NSHE * HHD * INTERD);
  unsigned short* WsuT  = allocb((size_t)NSHE * HHD * INTERD);
  unsigned short* WsdT  = allocb((size_t)NSHE * INTERD * HHD);
  unsigned short* WegT  = allocb((size_t)NRTE * HHD * INTERD);
  unsigned short* WeuT  = allocb((size_t)NRTE * HHD * INTERD);
  unsigned short* WedT  = allocb((size_t)NRTE * INTERD * HHD);
  // bf16 activations
  unsigned short* h_normb   = allocb((size_t)TT * HHD);
  unsigned short* cqb       = allocb((size_t)TT * QCC);
  unsigned short* qflatb    = allocb((size_t)TT * QPROJ);
  unsigned short* kvcnb     = allocb((size_t)TT * KVCC);
  unsigned short* kropeb    = allocb((size_t)TT * ROPED);
  unsigned short* kvflatb   = allocb((size_t)TT * KVPROJ);
  unsigned short* attnb     = allocb((size_t)BB * NHEADS * SS * SS);
  unsigned short* vT        = allocb((size_t)BB * NHEADS * VHD * SS);
  unsigned short* attn_catb = allocb((size_t)TT * HHD);
  unsigned short* tnormb    = allocb((size_t)TT * HHD);
  unsigned short* hidb      = allocb((size_t)TT * INTERD);
  // fp32 buffers
  float* cq_pre    = allocf((size_t)TT * QCC);
  float* ckv       = allocf((size_t)TT * CKVD);
  float* scores    = allocf((size_t)BB * NHEADS * SS * SS);
  float* attn_proj = allocf((size_t)TT * HHD);
  float* x1        = allocf((size_t)TT * HHD);
  float* hid_g     = allocf((size_t)TT * INTERD);
  float* hid_u     = allocf((size_t)TT * INTERD);
  float* ffn_acc   = allocf((size_t)TT * HHD);
  float* etmp      = allocf((size_t)TT * HHD);
  float* rweights  = allocf((size_t)TT * NRTE);

  const dim3 blk(256);
  auto ggrid = [](int M, int N) { return dim3(N / 64, M / 128); };
  auto tgrid = [](int K, int N, int z) { return dim3(N / 32, K / 32, z); };

  // ---- one-shot weight transpose + bf16 convert
  transpose_w_kernel<<<tgrid(HHD, QCC, 1),     blk, 0, stream>>>(W_dq,   WdqT,  HHD,   QCC);
  transpose_w_kernel<<<tgrid(QCC, QPROJ, 1),   blk, 0, stream>>>(W_uq,   WuqT,  QCC,   QPROJ);
  transpose_w_kernel<<<tgrid(HHD, CKVD, 1),    blk, 0, stream>>>(W_dkv,  WdkvT, HHD,   CKVD);
  transpose_w_kernel<<<tgrid(KVCC, KVPROJ, 1), blk, 0, stream>>>(W_ukv,  WukvT, KVCC,  KVPROJ);
  transpose_w_kernel<<<tgrid(HHD, HHD, 1),     blk, 0, stream>>>(W_o,    WoT,   HHD,   HHD);
  transpose_w_kernel<<<tgrid(HHD, INTERD, NSHE),   blk, 0, stream>>>(Ws_gate, WsgT, HHD, INTERD);
  transpose_w_kernel<<<tgrid(HHD, INTERD, NSHE),   blk, 0, stream>>>(Ws_up,   WsuT, HHD, INTERD);
  transpose_w_kernel<<<tgrid(INTERD, HHD, NSHE),   blk, 0, stream>>>(Ws_down, WsdT, INTERD, HHD);
  transpose_w_kernel<<<tgrid(HHD, INTERD, NRTE),   blk, 0, stream>>>(We_gate, WegT, HHD, INTERD);
  transpose_w_kernel<<<tgrid(HHD, INTERD, NRTE),   blk, 0, stream>>>(We_up,   WeuT, HHD, INTERD);
  transpose_w_kernel<<<tgrid(INTERD, HHD, NRTE),   blk, 0, stream>>>(We_down, WedT, INTERD, HHD);

  // ---- attention input norm + low-rank projections
  rmsnorm_kernel<<<TT, blk, 0, stream>>>(x, HHD, w_attn, h_normb, HHD, HHD);

  gemm_bb<false,false><<<ggrid(TT, QCC), blk, 0, stream>>>(
      h_normb, HHD, WdqT, HHD, cq_pre, QCC, TT, QCC, HHD);
  rmsnorm_kernel<<<TT, blk, 0, stream>>>(cq_pre, QCC, q_c_norm, cqb, QCC, QCC);
  gemm_bb<true,false><<<ggrid(TT, QPROJ), blk, 0, stream>>>(
      cqb, QCC, WuqT, QCC, qflatb, QPROJ, TT, QPROJ, QCC);

  gemm_bb<false,false><<<ggrid(TT, CKVD), blk, 0, stream>>>(
      h_normb, HHD, WdkvT, HHD, ckv, CKVD, TT, CKVD, HHD);
  rmsnorm_kernel<<<TT, blk, 0, stream>>>(ckv, CKVD, kv_c_norm, kvcnb, KVCC, KVCC);

  rope_q_kernel<<<(TT * NHEADS * 32) / 256, blk, 0, stream>>>(qflatb, freq);
  rope_k_kernel<<<(TT * 32) / 256, blk, 0, stream>>>(ckv, kropeb, freq);

  gemm_bb<true,false><<<ggrid(TT, KVPROJ), blk, 0, stream>>>(
      kvcnb, KVCC, WukvT, KVCC, kvflatb, KVPROJ, TT, KVPROJ, KVCC);

  vtranspose_kernel<<<dim3(VHD / 32, SS / 32, BB * NHEADS), blk, 0, stream>>>(kvflatb, vT);

  // ---- attention scores (natural [N][K] = KV-cache layout)
  for (int z = 0; z < BB * NHEADS; ++z) {
    int b = z / NHEADS, hh = z % NHEADS;
    float* sc = scores + (size_t)z * SS * SS;
    gemm_bb<false,false><<<ggrid(SS, SS), blk, 0, stream>>>(
        qflatb + (size_t)b * SS * QPROJ + hh * QDIM, QPROJ,
        kvflatb + (size_t)b * SS * KVPROJ + hh * KVD, KVPROJ,
        sc, SS, SS, SS, NOPED);
    gemm_bb<false,true><<<ggrid(SS, SS), blk, 0, stream>>>(
        qflatb + (size_t)b * SS * QPROJ + hh * QDIM + NOPED, QPROJ,
        kropeb + (size_t)b * SS * ROPED, ROPED,
        sc, SS, SS, SS, ROPED);
  }

  const float scale = 1.0f / sqrtf((float)QDIM);
  softmax_kernel<<<(long)BB * NHEADS * SS, blk, 0, stream>>>(scores, mask, attnb, SS, scale);

  // ---- attn @ v -> bf16 [b, s, h, d]
  for (int z = 0; z < BB * NHEADS; ++z) {
    int b = z / NHEADS, hh = z % NHEADS;
    gemm_bb<true,false><<<ggrid(SS, VHD), blk, 0, stream>>>(
        attnb + (size_t)z * SS * SS, SS,
        vT + (size_t)z * VHD * SS, SS,
        attn_catb + (size_t)b * SS * HHD + hh * VHD, HHD,
        SS, VHD, SS);
  }

  // ---- output projection + residual
  gemm_bb<false,false><<<ggrid(TT, HHD), blk, 0, stream>>>(
      attn_catb, HHD, WoT, HHD, attn_proj, HHD, TT, HHD, HHD);
  add_kernel<<<cdiv((long)TT * HHD, 256), blk, 0, stream>>>(x, attn_proj, x1, (long)TT * HHD);

  // ---- FFN norm
  rmsnorm_kernel<<<TT, blk, 0, stream>>>(x1, HHD, w_ffn, tnormb, HHD, HHD);

  // ---- shared experts (accumulate into ffn_acc)
  for (int e = 0; e < NSHE; ++e) {
    gemm_bb<false,false><<<ggrid(TT, INTERD), blk, 0, stream>>>(
        tnormb, HHD, WsgT + (size_t)e * HHD * INTERD, HHD, hid_g, INTERD, TT, INTERD, HHD);
    gemm_bb<false,false><<<ggrid(TT, INTERD), blk, 0, stream>>>(
        tnormb, HHD, WsuT + (size_t)e * HHD * INTERD, HHD, hid_u, INTERD, TT, INTERD, HHD);
    silu_mul_kernel<<<cdiv((long)TT * INTERD, 256), blk, 0, stream>>>(
        hid_g, hid_u, hidb, (long)TT * INTERD);
    if (e == 0)
      gemm_bb<false,false><<<ggrid(TT, HHD), blk, 0, stream>>>(
          hidb, INTERD, WsdT + (size_t)e * INTERD * HHD, INTERD, ffn_acc, HHD, TT, HHD, INTERD);
    else
      gemm_bb<false,true><<<ggrid(TT, HHD), blk, 0, stream>>>(
          hidb, INTERD, WsdT + (size_t)e * INTERD * HHD, INTERD, ffn_acc, HHD, TT, HHD, INTERD);
  }

  // ---- router + routed experts
  router_kernel<<<TT, blk, 0, stream>>>(tnormb, W_r, rweights);
  for (int e = 0; e < NRTE; ++e) {
    gemm_bb<false,false><<<ggrid(TT, INTERD), blk, 0, stream>>>(
        tnormb, HHD, WegT + (size_t)e * HHD * INTERD, HHD, hid_g, INTERD, TT, INTERD, HHD);
    gemm_bb<false,false><<<ggrid(TT, INTERD), blk, 0, stream>>>(
        tnormb, HHD, WeuT + (size_t)e * HHD * INTERD, HHD, hid_u, INTERD, TT, INTERD, HHD);
    silu_mul_kernel<<<cdiv((long)TT * INTERD, 256), blk, 0, stream>>>(
        hid_g, hid_u, hidb, (long)TT * INTERD);
    gemm_bb<false,false><<<ggrid(TT, HHD), blk, 0, stream>>>(
        hidb, INTERD, WedT + (size_t)e * INTERD * HHD, INTERD, etmp, HHD, TT, HHD, INTERD);
    axpy_rows_kernel<<<cdiv((long)TT * HHD, 256), blk, 0, stream>>>(
        ffn_acc, etmp, rweights, e, (long)TT * HHD, HHD);
  }

  // ---- final residual
  add_kernel<<<cdiv((long)TT * HHD, 256), blk, 0, stream>>>(
      x1, ffn_acc, out, (long)TT * HHD);
}
